// MLA_63359357551389
// MI455X (gfx1250) — matrix operand order repaired and checked by reading, not compiled, not run on hardware
//
#include <hip/hip_runtime.h>

// ---------------- problem constants ----------------
#define DIMX   2048
#define NH     16
#define QLORA  1536
#define KVLORA 512
#define NOPED  128
#define ROPED  64
#define QKD    192      // NOPE + ROPE
#define VDIM   128
#define BB     2
#define SSEQ   2048
#define BS     (BB * SSEQ)   // 4096 tokens

typedef __bf16 bf16;
typedef __attribute__((ext_vector_type(16))) __bf16 bf16x16;
typedef __attribute__((ext_vector_type(8)))  __bf16 bf16x8;
typedef __attribute__((ext_vector_type(8)))  float  f32x8;
typedef __attribute__((ext_vector_type(4)))  unsigned int u32x4;
typedef __attribute__((ext_vector_type(8)))  int i32x8;
typedef __attribute__((ext_vector_type(4)))  int i32x4;

// ---------------- WMMA fragment helpers (CDNA5 16x16x32 bf16 layouts) ----------------
// A 16x32 (MxK): lane L (m=L%16, h=L/16): elems 0..7 -> K=8h+0..7 ; elems 8..15 -> K=16+8h+0..7
static __device__ __forceinline__ bf16x16 make_a16(const bf16* p) {
  bf16x8 lo = *(const bf16x8*)p;
  bf16x8 hi = *(const bf16x8*)(p + 16);
  return __builtin_shufflevector(lo, hi, 0,1,2,3,4,5,6,7,8,9,10,11,12,13,14,15);
}

// B operand: elems 0..7 -> K=8h+0..7 ; elems 8..15 -> K=16+8h+0..7 (lane n=L%16 row, h=L/16)
static __device__ __forceinline__ bf16x16 make_b16(const bf16* p) {
  bf16x8 lo = *(const bf16x8*)p;
  bf16x8 hi = *(const bf16x8*)(p + 16);
  return __builtin_shufflevector(lo, hi, 0,1,2,3,4,5,6,7,8,9,10,11,12,13,14,15);
}

static __device__ __forceinline__ bf16x16 load_a_frag(const bf16* A, int lda) {
  const int lane = threadIdx.x & 31;
  const int m = lane & 15, h = lane >> 4;
  return make_a16(A + (size_t)m * lda + 8 * h);
}

static __device__ __forceinline__ bf16x16 load_b_frag(const bf16* W, int ldw) {
  const int lane = threadIdx.x & 31;
  const int n = lane & 15, h = lane >> 4;
  return make_b16(W + (size_t)n * ldw + 8 * h);
}

static __device__ __forceinline__ f32x8 wmma_bf16(bf16x16 a, bf16x16 b, f32x8 c) {
  return __builtin_amdgcn_wmma_f32_16x16x32_bf16(false, a, false, b, (short)0, c, false, false);
}

// Low 32 bits of the generic address of an LDS object = LDS byte offset
static __device__ __forceinline__ unsigned lds_off(const void* p) {
  return (unsigned)(uintptr_t)p;
}

// ---------------- Tensor Data Mover: 2D tile global -> LDS ----------------
// Packs a D# per CDNA5 ISA ch.8.3/8.4: group0 = {count/type/lds/global addr},
// group1 = {data_size=2B, tensor dims, tile dims, dim0 stride}. 2D tile so
// groups 2/3 (and the trailing group) are zero. Issued by one wave;
// completion tracked via TENSORcnt.
static __device__ __forceinline__ void tdm_load_2d(const bf16* gsrc, unsigned lds_byte_off,
                                                   unsigned tensor_rows, unsigned stride_elems,
                                                   unsigned tile_rows, unsigned tile_cols) {
  unsigned long long ga = (unsigned long long)(uintptr_t)gsrc;
  u32x4 g0 = { 1u,                                     // count=1 valid descriptor
               lds_byte_off,                           // lds_addr
               (unsigned)ga,                           // global_addr[31:0]
               (unsigned)((ga >> 32) & 0x01ffffffu) | 0x80000000u };  // addr[56:32] | type=2
  i32x8 g1;
  g1[0] = (int)(1u << 16);                                              // data_size = 2 bytes
  g1[1] = (int)((stride_elems & 0xffffu) << 16);                        // tensor_dim0[15:0]
  g1[2] = (int)(((stride_elems >> 16) & 0xffffu) | ((tensor_rows & 0xffffu) << 16));
  g1[3] = (int)(((tensor_rows >> 16) & 0xffffu) | ((tile_cols & 0xffffu) << 16)); // tile_dim0
  g1[4] = (int)(tile_rows & 0xffffu);                                   // tile_dim1, tile_dim2=0
  g1[5] = (int)stride_elems;                                            // tensor_dim0_stride lo
  g1[6] = 0;
  g1[7] = 0;
  i32x4 gz4 = { 0, 0, 0, 0 };
  i32x8 gz8 = { 0, 0, 0, 0, 0, 0, 0, 0 };
  __builtin_amdgcn_tensor_load_to_lds(g0, g1, gz4, gz4, gz8, 0);
}

// ---------------- generic f32 -> bf16 convert ----------------
__global__ void cvt_f32_bf16_kernel(const float* __restrict__ x, bf16* __restrict__ y, int n) {
  int i = blockIdx.x * 256 + threadIdx.x;
  if (i < n) y[i] = (bf16)x[i];
}

// ---------------- WMMA GEMM: C[M,N] = A[M,K] * W[N,K]^T ----------------
// 4 waves / block, block tile 128(M) x 64(N), K-step 32.
// TDM double-buffered LDS staging: wave0 issues A(128x32) + W(64x32) tile DMAs.
#define GEMM_BOFF (128 * 32)          // B tile offset (elements) inside one buffer
#define GEMM_BUFE ((128 + 64) * 32)   // elements per buffer

template<bool OUT_BF16>
__global__ __launch_bounds__(128) void gemm_wmma_kernel(
    const bf16* __restrict__ A, const bf16* __restrict__ W,
    void* __restrict__ Cout, int M, int N, int K) {
  const int col0 = blockIdx.x * 64;
  const int row0 = blockIdx.y * 128;
  const int wid  = threadIdx.x >> 5;
  const int lane = threadIdx.x & 31;
  const int ln = lane & 15, lh = lane >> 4;

  // [buffer][ A 128x32 | B 64x32 ] bf16, 24 KB total
  __shared__ alignas(16) bf16 smem[2][GEMM_BUFE];

  if (wid == 0) {
    tdm_load_2d(A + (size_t)row0 * K, lds_off(&smem[0][0]),         128, (unsigned)K, 128, 32);
    tdm_load_2d(W + (size_t)col0 * K, lds_off(&smem[0][GEMM_BOFF]), 64,  (unsigned)K, 64,  32);
  }

  f32x8 acc[2][4] = {};
  int p = 0;
  for (int k0 = 0; k0 < K; k0 += 32) {
    if (wid == 0) __builtin_amdgcn_s_wait_tensorcnt(0);  // buffer p ready
    __syncthreads();
    if (wid == 0 && (k0 + 32) < K) {                     // prefetch next tile into p^1
      tdm_load_2d(A + (size_t)row0 * K + (k0 + 32), lds_off(&smem[p ^ 1][0]),
                  128, (unsigned)K, 128, 32);
      tdm_load_2d(W + (size_t)col0 * K + (k0 + 32), lds_off(&smem[p ^ 1][GEMM_BOFF]),
                  64, (unsigned)K, 64, 32);
    }
    bf16x16 a[2], b[4];
#pragma unroll
    for (int mi = 0; mi < 2; ++mi)
      a[mi] = make_a16(&smem[p][(32 * wid + 16 * mi + ln) * 32 + 8 * lh]);
#pragma unroll
    for (int ni = 0; ni < 4; ++ni)
      b[ni] = make_b16(&smem[p][GEMM_BOFF + (16 * ni + ln) * 32 + 8 * lh]);
#pragma unroll
    for (int mi = 0; mi < 2; ++mi)
#pragma unroll
      for (int ni = 0; ni < 4; ++ni)
        acc[mi][ni] = wmma_bf16(a[mi], b[ni], acc[mi][ni]);
    __syncthreads();
    p ^= 1;
  }

  // D layout: lane L: n = L%16, rows m = 8*(L/16) + r
  const int rbase = row0 + 32 * wid;
#pragma unroll
  for (int mi = 0; mi < 2; ++mi)
#pragma unroll
    for (int ni = 0; ni < 4; ++ni) {
      size_t base = (size_t)(rbase + 16 * mi + 8 * lh) * N + (col0 + 16 * ni + ln);
#pragma unroll
      for (int r = 0; r < 8; ++r) {
        size_t idx = base + (size_t)r * N;
        if (OUT_BF16) ((bf16*)Cout)[idx] = (bf16)acc[mi][ni][r];
        else          ((float*)Cout)[idx] = acc[mi][ni][r];
      }
    }
}

// ---------------- RMSNorm (f32 in -> bf16 out), one block per row ----------------
__global__ __launch_bounds__(256) void rmsnorm_kernel(
    const float* __restrict__ X, const float* __restrict__ w,
    bf16* __restrict__ Y, int C) {
  const int row = blockIdx.x;
  const float* x = X + (size_t)row * C;
  bf16* y = Y + (size_t)row * C;
  float ss = 0.f;
  for (int c = threadIdx.x; c < C; c += 256) { float v = x[c]; ss += v * v; }
  __shared__ float sm[8];
#pragma unroll
  for (int m = 16; m >= 1; m >>= 1) ss += __shfl_xor(ss, m);
  if ((threadIdx.x & 31) == 0) sm[threadIdx.x >> 5] = ss;
  __syncthreads();
  if (threadIdx.x < 8) {
    float v = sm[threadIdx.x];
#pragma unroll
    for (int m = 4; m >= 1; m >>= 1) v += __shfl_xor(v, m);
    if (threadIdx.x == 0) sm[0] = v;
  }
  __syncthreads();
  const float r = rsqrtf(sm[0] / (float)C + 1e-6f);
  for (int c = threadIdx.x; c < C; c += 256) y[c] = (bf16)(x[c] * r * w[c]);
}

// ---------------- RoPE on q + relayout to head-major Q [B*H][S][192] bf16 ----------------
__global__ void rope_q_kernel(const bf16* __restrict__ qf,
                              const float* __restrict__ cosT, const float* __restrict__ sinT,
                              bf16* __restrict__ Qh) {
  int idx = blockIdx.x * 256 + threadIdx.x;            // over BS*NH*QKD
  if (idx >= BS * NH * QKD) return;
  int d = idx % QKD;
  int th = idx / QKD;
  int h = th % NH;
  int token = th / NH;
  int s = token & (SSEQ - 1);
  int b = token >> 11;
  const bf16* qrow = qf + (size_t)token * (NH * QKD) + h * QKD;
  float v;
  if (d < NOPED) {
    v = (float)qrow[d];
  } else {
    int dd = d - NOPED;                                 // 0..63
    float xv = (float)qrow[d];
    float rot = (dd < 32) ? -(float)qrow[NOPED + dd + 32]
                          :  (float)qrow[NOPED + dd - 32];
    v = xv * cosT[s * ROPED + dd] + rot * sinT[s * ROPED + dd];
  }
  Qh[((size_t)(b * NH + h) * SSEQ + s) * QKD + d] = (bf16)v;
}

// ---------------- kv post: rmsnorm(kv_c) -> bf16, RoPE(k_pe) -> f32 ----------------
__global__ __launch_bounds__(256) void kv_post_kernel(
    const float* __restrict__ kvf, const float* __restrict__ kvw,
    const float* __restrict__ cosT, const float* __restrict__ sinT,
    bf16* __restrict__ kvcn, float* __restrict__ kpe) {
  const int token = blockIdx.x;
  const float* row = kvf + (size_t)token * (KVLORA + ROPED);
  float ss = 0.f;
  for (int c = threadIdx.x; c < KVLORA; c += 256) { float v = row[c]; ss += v * v; }
  __shared__ float sm[8];
#pragma unroll
  for (int m = 16; m >= 1; m >>= 1) ss += __shfl_xor(ss, m);
  if ((threadIdx.x & 31) == 0) sm[threadIdx.x >> 5] = ss;
  __syncthreads();
  if (threadIdx.x < 8) {
    float v = sm[threadIdx.x];
#pragma unroll
    for (int m = 4; m >= 1; m >>= 1) v += __shfl_xor(v, m);
    if (threadIdx.x == 0) sm[0] = v;
  }
  __syncthreads();
  const float r = rsqrtf(sm[0] / (float)KVLORA + 1e-6f);
  for (int c = threadIdx.x; c < KVLORA; c += 256)
    kvcn[(size_t)token * KVLORA + c] = (bf16)(row[c] * r * kvw[c]);
  if (threadIdx.x < ROPED) {
    int dd = threadIdx.x;
    int s = token & (SSEQ - 1);
    float xv = row[KVLORA + dd];
    float rot = (dd < 32) ? -row[KVLORA + dd + 32] : row[KVLORA + dd - 32];
    kpe[(size_t)token * ROPED + dd] = xv * cosT[s * ROPED + dd] + rot * sinT[s * ROPED + dd];
  }
}

// ---------------- build K [B*H][S][192] and V^T [B*H][128][S] ----------------
__global__ void build_kv_kernel(const bf16* __restrict__ kvb, const float* __restrict__ kpe,
                                bf16* __restrict__ Kh, bf16* __restrict__ VT) {
  int idx = blockIdx.x * 256 + threadIdx.x;             // over BS*NH*320
  if (idx >= BS * NH * 320) return;
  int c = idx % 320;
  int th = idx / 320;
  int h = th % NH;
  int token = th / NH;
  int s = token & (SSEQ - 1);
  int b = token >> 11;
  int bh = b * NH + h;
  if (c < NOPED) {
    Kh[((size_t)bh * SSEQ + s) * QKD + c] = kvb[(size_t)token * (NH * 256) + h * 256 + c];
  } else if (c < 256) {
    int j = c - NOPED;
    VT[((size_t)bh * VDIM + j) * SSEQ + s] = kvb[(size_t)token * (NH * 256) + h * 256 + c];
  } else {
    int dd = c - 256;                                   // 0..63
    Kh[((size_t)bh * SSEQ + s) * QKD + NOPED + dd] = (bf16)kpe[(size_t)token * ROPED + dd];
  }
}

// ---------------- causal flash attention, WMMA, per-wave 16 q-rows ----------------
__global__ __launch_bounds__(128) void attn_kernel(
    const bf16* __restrict__ Qh, const bf16* __restrict__ Kh,
    const bf16* __restrict__ VT, bf16* __restrict__ attn) {
  const int bh   = blockIdx.x;              // b*NH + h
  const int wid  = threadIdx.x >> 5;
  const int lane = threadIdx.x & 31;
  const int lm   = lane & 15, lh = lane >> 4;
  const int q0   = blockIdx.y * 64 + wid * 16;
  const float scale = 0.0721687836f;        // 192^-0.5

  __shared__ alignas(16) __bf16 pshare[4][16][40];

  const bf16* Q = Qh + (size_t)bh * SSEQ * QKD;
  const bf16* K = Kh + (size_t)bh * SSEQ * QKD;
  const bf16* V = VT + (size_t)bh * VDIM * SSEQ;

  bf16x16 aq[6];
#pragma unroll
  for (int ks = 0; ks < 6; ++ks)
    aq[ks] = load_a_frag(Q + (size_t)q0 * QKD + ks * 32, QKD);

  f32x8 o[8] = {};
  float mrow[8], lrow[8];
#pragma unroll
  for (int r = 0; r < 8; ++r) { mrow[r] = -1e30f; lrow[r] = 0.f; }

  for (int k0 = 0; k0 < q0 + 16; k0 += 32) {
    __builtin_prefetch(K + (size_t)(k0 + 32) * QKD, 0, 3);
    // ---- S = Q K^T for this 16x32 tile ----
    f32x8 s0 = {}, s1 = {};
#pragma unroll
    for (int ks = 0; ks < 6; ++ks) {
      bf16x16 b0 = load_b_frag(K + (size_t)k0 * QKD + ks * 32, QKD);
      bf16x16 b1 = load_b_frag(K + (size_t)(k0 + 16) * QKD + ks * 32, QKD);
      s0 = wmma_bf16(aq[ks], b0, s0);
      s1 = wmma_bf16(aq[ks], b1, s1);
    }
    // ---- scale + causal mask ----
    float sv0[8], sv1[8];
#pragma unroll
    for (int r = 0; r < 8; ++r) {
      int qi = q0 + 8 * lh + r;
      sv0[r] = (k0 + lm      <= qi) ? s0[r] * scale : -1e30f;
      sv1[r] = (k0 + 16 + lm <= qi) ? s1[r] * scale : -1e30f;
    }
    // ---- row max (row = 8*lh + r lives in one 16-lane group) ----
    float tmax[8];
#pragma unroll
    for (int r = 0; r < 8; ++r) tmax[r] = fmaxf(sv0[r], sv1[r]);
#pragma unroll
    for (int m = 1; m <= 8; m <<= 1)
#pragma unroll
      for (int r = 0; r < 8; ++r) tmax[r] = fmaxf(tmax[r], __shfl_xor(tmax[r], m));
    float alpha[8];
#pragma unroll
    for (int r = 0; r < 8; ++r) {
      float mn = fmaxf(mrow[r], tmax[r]);
      alpha[r] = __expf(mrow[r] - mn);
      mrow[r] = mn;
    }
    // ---- P = exp(S - m), row sum ----
    float psum[8];
#pragma unroll
    for (int r = 0; r < 8; ++r) {
      float p0 = __expf(sv0[r] - mrow[r]);
      float p1 = __expf(sv1[r] - mrow[r]);
      sv0[r] = p0; sv1[r] = p1;
      psum[r] = p0 + p1;
    }
#pragma unroll
    for (int m = 1; m <= 8; m <<= 1)
#pragma unroll
      for (int r = 0; r < 8; ++r) psum[r] += __shfl_xor(psum[r], m);
#pragma unroll
    for (int r = 0; r < 8; ++r) lrow[r] = lrow[r] * alpha[r] + psum[r];
    // ---- rescale O ----
#pragma unroll
    for (int jf = 0; jf < 8; ++jf)
#pragma unroll
      for (int r = 0; r < 8; ++r) o[jf][r] *= alpha[r];
    // ---- relayout P (D-layout -> A-layout) through per-wave LDS ----
#pragma unroll
    for (int r = 0; r < 8; ++r) {
      pshare[wid][8 * lh + r][lm]      = (__bf16)sv0[r];
      pshare[wid][8 * lh + r][16 + lm] = (__bf16)sv1[r];
    }
    asm volatile("s_wait_dscnt 0x0" ::: "memory");
    bf16x16 ap = make_a16(&pshare[wid][lm][8 * lh]);
    // ---- O += P * V ----
#pragma unroll
    for (int jf = 0; jf < 8; ++jf) {
      bf16x16 bv = load_b_frag(V + (size_t)(jf * 16) * SSEQ + k0, SSEQ);
      o[jf] = wmma_bf16(ap, bv, o[jf]);
    }
  }
  // ---- epilogue: O / l, scatter to attn [token][h*128 + j] ----
  const int b = bh >> 4, h = bh & (NH - 1);
  float inv[8];
#pragma unroll
  for (int r = 0; r < 8; ++r) inv[r] = 1.0f / lrow[r];
#pragma unroll
  for (int jf = 0; jf < 8; ++jf)
#pragma unroll
    for (int r = 0; r < 8; ++r) {
      int s = q0 + 8 * lh + r;
      attn[(size_t)(b * SSEQ + s) * (NH * VDIM) + h * VDIM + jf * 16 + lm] =
          (bf16)(o[jf][r] * inv[r]);
    }
}

// ---------------- host launch ----------------
static inline int cdiv(int a, int b) { return (a + b - 1) / b; }

extern "C" void kernel_launch(void* const* d_in, const int* in_sizes, int n_in,
                              void* d_out, int out_size, void* d_ws, size_t ws_size,
                              hipStream_t stream) {
  (void)in_sizes; (void)n_in; (void)out_size; (void)ws_size;
  const float* x    = (const float*)d_in[0];
  const float* cosT = (const float*)d_in[1];
  const float* sinT = (const float*)d_in[2];
  const float* wq_a = (const float*)d_in[3];
  const float* qnw  = (const float*)d_in[4];
  const float* wq_b = (const float*)d_in[5];
  const float* wkva = (const float*)d_in[6];
  const float* kvnw = (const float*)d_in[7];
  const float* wkvb = (const float*)d_in[8];
  const float* wo   = (const float*)d_in[9];

  char* ws = (char*)d_ws;
  size_t off = 0;
  auto alloc = [&](size_t bytes) -> void* {
    void* p = ws + off;
    off = (off + bytes + 255) & ~(size_t)255;
    return p;
  };

  bf16*  xb      = (bf16*)alloc((size_t)BS * DIMX * 2);
  bf16*  wqa_b   = (bf16*)alloc((size_t)QLORA * DIMX * 2);
  bf16*  wqb_b   = (bf16*)alloc((size_t)NH * QKD * QLORA * 2);
  bf16*  wkva_b  = (bf16*)alloc((size_t)(KVLORA + ROPED) * DIMX * 2);
  bf16*  wkvb_b  = (bf16*)alloc((size_t)NH * 256 * KVLORA * 2);
  bf16*  wo_b    = (bf16*)alloc((size_t)DIMX * NH * VDIM * 2);
  float* qlora   = (float*)alloc((size_t)BS * QLORA * 4);
  bf16*  qlora_n = (bf16*)alloc((size_t)BS * QLORA * 2);
  bf16*  qf      = (bf16*)alloc((size_t)BS * NH * QKD * 2);
  bf16*  Qh      = (bf16*)alloc((size_t)BB * NH * SSEQ * QKD * 2);
  float* kvf     = (float*)alloc((size_t)BS * (KVLORA + ROPED) * 4);
  bf16*  kvcn    = (bf16*)alloc((size_t)BS * KVLORA * 2);
  float* kpe     = (float*)alloc((size_t)BS * ROPED * 4);
  bf16*  kvb     = (bf16*)alloc((size_t)BS * NH * 256 * 2);
  bf16*  Kh      = (bf16*)alloc((size_t)BB * NH * SSEQ * QKD * 2);
  bf16*  VT      = (bf16*)alloc((size_t)BB * NH * VDIM * SSEQ * 2);
  bf16*  attn    = (bf16*)alloc((size_t)BS * NH * VDIM * 2);

  // --- converts ---
  auto cvt = [&](const float* src, bf16* dst, int n) {
    cvt_f32_bf16_kernel<<<cdiv(n, 256), 256, 0, stream>>>(src, dst, n);
  };
  cvt(x,    xb,     BS * DIMX);
  cvt(wq_a, wqa_b,  QLORA * DIMX);
  cvt(wq_b, wqb_b,  NH * QKD * QLORA);
  cvt(wkva, wkva_b, (KVLORA + ROPED) * DIMX);
  cvt(wkvb, wkvb_b, NH * 256 * KVLORA);
  cvt(wo,   wo_b,   DIMX * NH * VDIM);

  // --- q path ---
  gemm_wmma_kernel<false><<<dim3(QLORA / 64, BS / 128), 128, 0, stream>>>(
      xb, wqa_b, qlora, BS, QLORA, DIMX);
  rmsnorm_kernel<<<BS, 256, 0, stream>>>(qlora, qnw, qlora_n, QLORA);
  gemm_wmma_kernel<true><<<dim3((NH * QKD) / 64, BS / 128), 128, 0, stream>>>(
      qlora_n, wqb_b, qf, BS, NH * QKD, QLORA);
  rope_q_kernel<<<cdiv(BS * NH * QKD, 256), 256, 0, stream>>>(qf, cosT, sinT, Qh);

  // --- kv path ---
  gemm_wmma_kernel<false><<<dim3((KVLORA + ROPED) / 64, BS / 128), 128, 0, stream>>>(
      xb, wkva_b, kvf, BS, KVLORA + ROPED, DIMX);
  kv_post_kernel<<<BS, 256, 0, stream>>>(kvf, kvnw, cosT, sinT, kvcn, kpe);
  gemm_wmma_kernel<true><<<dim3((NH * 256) / 64, BS / 128), 128, 0, stream>>>(
      kvcn, wkvb_b, kvb, BS, NH * 256, KVLORA);
  build_kv_kernel<<<cdiv(BS * NH * 320, 256), 256, 0, stream>>>(kvb, kpe, Kh, VT);

  // --- attention ---
  attn_kernel<<<dim3(BB * NH, SSEQ / 64), 128, 0, stream>>>(Qh, Kh, VT, attn);

  // --- output projection ---
  gemm_wmma_kernel<false><<<dim3(DIMX / 64, BS / 128), 128, 0, stream>>>(
      attn, wo_b, (float*)d_out, BS, DIMX, DIMX);
}
